// NGmeet_20409684590803
// MI455X (gfx1250) — compile-verified, matrix operand order checked by reading, not run on hardware
//
#include <hip/hip_runtime.h>

typedef __attribute__((ext_vector_type(16))) __bf16 v16bf;
typedef __attribute__((ext_vector_type(8)))  float  v8f;

#define NB    4
#define BAND  128
#define HH    256
#define WW    256
#define HWD   65536L
#define PADK  16          // padded subspace dim (k_sub <= 14)

__device__ inline v8f zero8() {
  v8f c;
#pragma unroll
  for (int r = 0; r < 8; ++r) c[r] = 0.0f;
  return c;
}

__device__ inline v8f wmma_bf16(v16bf a, v16bf b, v8f c) {
  // v_wmma_f32_16x16x32_bf16 : D = A(16x32) * B(32x16) + C
  return __builtin_amdgcn_wmma_f32_16x16x32_bf16(false, a, false, b, (short)0, c,
                                                 false, false);
}

// A-style fragment from a row-major MxK matrix (rows row0..row0+15).
// lane: m = lane&15, kbase = (lane>>4)*8 ; element j: K = kb + (j&7) + (j>=8)*16
__device__ inline v16bf frag_mk(const float* __restrict__ p, long ld, int row0,
                                int k0, int lane) {
  int m = lane & 15;
  int kb = (lane >> 4) << 3;
  const float* r = p + (long)(row0 + m) * ld + (k0 + kb);
  v16bf f;
#pragma unroll
  for (int j = 0; j < 16; ++j) {
    int K = (j & 7) + ((j >> 3) << 4);
    f[j] = (__bf16)r[K];
  }
  return f;
}

// B-style fragment from a row-major KxN matrix: lane n = lane&15.
// kmax guards reads past the stored K extent (returns 0 -> zero-pad).
__device__ inline v16bf frag_kn(const float* __restrict__ p, long ld, int k0,
                                int col0, int lane, int kmax, float scale) {
  int n = lane & 15;
  int kb = (lane >> 4) << 3;
  v16bf f;
#pragma unroll
  for (int j = 0; j < 16; ++j) {
    int K = kb + (j & 7) + ((j >> 3) << 4);
    int kk = k0 + K;
    float v = (kk < kmax) ? p[(long)kk * ld + (col0 + n)] : 0.0f;
    f[j] = (__bf16)(v * scale);
  }
  return f;
}

// ---------------------------------------------------------------- utility
__global__ void zero_kernel(float* __restrict__ p, long n) {
  long i = (long)blockIdx.x * blockDim.x + threadIdx.x;
  for (; i < n; i += (long)gridDim.x * blockDim.x) p[i] = 0.0f;
}

__global__ void init_eimg_kernel(const float* __restrict__ Nimg,
                                 float* __restrict__ Eimg) {
  long n = (long)NB * BAND * HWD;
  long i = (long)blockIdx.x * blockDim.x + threadIdx.x;
  for (; i < n; i += (long)gridDim.x * blockDim.x) Eimg[i] = Nimg[i] * 255.0f;
}

__global__ void rho_reduce_kernel(const float* __restrict__ inW,
                                  float* __restrict__ scal) {
  __shared__ float rs[256];
  int b = blockIdx.y;
  long n = (long)BAND * HWD;
  float s = 0.0f;
  for (long i = (long)blockIdx.x * blockDim.x + threadIdx.x; i < n;
       i += (long)gridDim.x * blockDim.x)
    s += inW[(long)b * n + i];
  rs[threadIdx.x] = s;
  __syncthreads();
  for (int st = 128; st > 0; st >>= 1) {
    if (threadIdx.x < st) rs[threadIdx.x] += rs[threadIdx.x + st];
    __syncthreads();
  }
  if (threadIdx.x == 0) atomicAdd(&scal[b], rs[0]);
}

__global__ void rho_final_kernel(float* __restrict__ scal) {
  if (threadIdx.x < NB)
    scal[threadIdx.x] = 9.0f * scal[threadIdx.x] / ((float)BAND * (float)HWD);
}

// ---------------------------------------------------------------- gram = Eimg * Eimg^T
// one wave per (16x16 output tile, K-chunk of 1024); f32 atomic accumulate
__global__ void gram_kernel(const float* __restrict__ Eimg,
                            float* __restrict__ gram) {
  int lane = threadIdx.x & 31;
  int wid = blockIdx.x * (blockDim.x >> 5) + (threadIdx.x >> 5); // 0..4095
  int b = blockIdx.y;
  int tile = wid >> 6;   // 0..63  (8x8 tiles)
  int kc = wid & 63;     // 64 chunks of 1024
  int tm = tile >> 3, tn = tile & 7;
  const float* Eb = Eimg + (long)b * BAND * HWD;
  v8f c = zero8();
  int k0 = kc << 10;
  for (int kk = 0; kk < 1024; kk += 32) {
    v16bf a = frag_mk(Eb, HWD, tm * 16, k0 + kk, lane);
    v16bf bb = frag_mk(Eb, HWD, tn * 16, k0 + kk, lane); // B[k][n] = Eimg[n][k]
    c = wmma_bf16(a, bb, c);
  }
  float* g = gram + b * BAND * BAND;
  int n = lane & 15, mh = (lane >> 4) << 3;
#pragma unroll
  for (int r = 0; r < 8; ++r)
    atomicAdd(&g[(tm * 16 + r + mh) * BAND + tn * 16 + n], c[r]);
}

// ---------------------------------------------------------------- 128x128 symmetric Jacobi
// one block (128 thr) per batch. A in LDS, V in global scratch.
// tournament pairing: 64 disjoint pairs/round, column phase then row phase.
__global__ void __launch_bounds__(128) jacobi_kernel(const float* __restrict__ gram,
                                                     float* __restrict__ Vg,
                                                     float* __restrict__ E, int ksub) {
  __shared__ float A[BAND * BAND];
  const int N = BAND;
  int b = blockIdx.x;
  int tid = threadIdx.x;
  float* Vb = Vg + (long)b * N * N;
  for (int i = tid; i < N * N; i += 128) {
    A[i] = gram[(long)b * N * N + i];
    Vb[i] = ((i / N) == (i % N)) ? 1.0f : 0.0f;
  }
  __syncthreads();

  int pairI = tid >> 1;
  int half = tid & 1;
  for (int sweep = 0; sweep < 8; ++sweep) {
    for (int round = 0; round < N - 1; ++round) {
      int s0 = pairI, s1 = N - 1 - pairI;
      int p = (s0 == 0) ? 0 : ((s0 - 1 + round) % (N - 1)) + 1;
      int q = ((s1 - 1 + round) % (N - 1)) + 1;
      if (p > q) { int t = p; p = q; q = t; }
      float app = A[p * N + p], aqq = A[q * N + q], apq = A[p * N + q];
      float c = 1.0f, s = 0.0f;
      if (fabsf(apq) > 1e-12f) {
        float th = (aqq - app) / (2.0f * apq);
        float t = ((th >= 0.0f) ? 1.0f : -1.0f) / (fabsf(th) + sqrtf(th * th + 1.0f));
        c = 1.0f / sqrtf(t * t + 1.0f);
        s = t * c;
      }
      __syncthreads();
      // column phase: A <- A*J, V <- V*J (disjoint column pairs)
      for (int r = half * 64; r < half * 64 + 64; ++r) {
        float arp = A[r * N + p], arq = A[r * N + q];
        A[r * N + p] = c * arp - s * arq;
        A[r * N + q] = s * arp + c * arq;
        float vrp = Vb[r * N + p], vrq = Vb[r * N + q];
        Vb[r * N + p] = c * vrp - s * vrq;
        Vb[r * N + q] = s * vrp + c * vrq;
      }
      __syncthreads();
      // row phase: A <- J^T*A (disjoint row pairs)
      for (int j = half * 64; j < half * 64 + 64; ++j) {
        float apj = A[p * N + j], aqj = A[q * N + j];
        A[p * N + j] = c * apj - s * aqj;
        A[q * N + j] = s * apj + c * aqj;
      }
      __syncthreads();
    }
  }
  // E = top-ksub eigvecs (desc), zero-padded 128x32 row-major
  for (int i = tid; i < N * 32; i += 128) E[(long)b * N * 32 + i] = 0.0f;
  __syncthreads();
  if (tid == 0) {
    for (int j = 0; j < ksub; ++j) {
      float best = -3.0e38f; int bi = 0;
      for (int m = 0; m < N; ++m) {
        float d = A[m * N + m];
        if (d > best) { best = d; bi = m; }
      }
      A[bi * N + bi] = -3.0e38f;
      for (int r = 0; r < N; ++r)
        E[((long)b * N + r) * 32 + j] = Vb[r * N + bi];
    }
  }
}

// ---------------------------------------------------------------- Eproj / Nproj = Et @ {Eimg, 255*Nimg}
__global__ void proj_kernel(const float* __restrict__ E,
                            const float* __restrict__ Eimg,
                            const float* __restrict__ Nimg,
                            float* __restrict__ Eproj, float* __restrict__ Nproj) {
  int lane = threadIdx.x & 31;
  int tn = blockIdx.x * (blockDim.x >> 5) + (threadIdx.x >> 5); // 0..4095
  int b = blockIdx.y;
  const float* Eb = E + (long)b * BAND * 32;
  const float* Ib = Eimg + (long)b * BAND * HWD;
  const float* Nb = Nimg + (long)b * BAND * HWD;
  v8f ce = zero8(), cn = zero8();
  for (int k0 = 0; k0 < BAND; k0 += 32) {
    v16bf a = frag_kn(Eb, 32, k0, 0, lane, BAND, 1.0f);     // Et[m][k] = E[k][m]
    v16bf be = frag_kn(Ib, HWD, k0, tn * 16, lane, BAND, 1.0f);
    v16bf bn = frag_kn(Nb, HWD, k0, tn * 16, lane, BAND, 255.0f);
    ce = wmma_bf16(a, be, ce);
    cn = wmma_bf16(a, bn, cn);
  }
  int n = tn * 16 + (lane & 15), mh = (lane >> 4) << 3;
#pragma unroll
  for (int r = 0; r < 8; ++r) {
    long o = ((long)b * PADK + (r + mh)) * HWD + n;
    Eproj[o] = ce[r];
    Nproj[o] = cn[r];
  }
}

// ---------------------------------------------------------------- per-band means + MSE(Eproj,Nproj)
__global__ void stats_kernel(const float* __restrict__ Eproj,
                             const float* __restrict__ Nproj,
                             float* __restrict__ scal) {
  __shared__ float rs[256], rq[256];
  int bi = blockIdx.x; // b*16 + band
  long base = (long)bi * HWD;
  float s = 0.0f, q = 0.0f;
  for (int i = threadIdx.x; i < (int)HWD; i += blockDim.x) {
    float e = Eproj[base + i];
    float d = e - Nproj[base + i];
    s += e; q += d * d;
  }
  rs[threadIdx.x] = s; rq[threadIdx.x] = q;
  __syncthreads();
  for (int st = 128; st > 0; st >>= 1) {
    if (threadIdx.x < st) { rs[threadIdx.x] += rs[threadIdx.x + st];
                            rq[threadIdx.x] += rq[threadIdx.x + st]; }
    __syncthreads();
  }
  if (threadIdx.x == 0) { scal[4 + bi] = rs[0]; atomicAdd(&scal[68], rq[0]); }
}

__global__ void finalize_kernel(float* __restrict__ scal,
                                const float* __restrict__ nSig, int ksub, int iter) {
  if (threadIdx.x == 0) {
    for (int i = 0; i < NB * PADK; ++i) scal[4 + i] *= (1.0f / (float)HWD);
    float msd = scal[68] / ((float)NB * (float)ksub * (float)HWD);
    float ns = nSig[0];
    float s1 = 0.54f * sqrtf(fabsf(ns * ns - msd));
    float sigr = (iter > 0) ? s1 : ns * sqrtf(6.0f / 128.0f);
    scal[69] = 5.0f * sqrtf(2.0f) * sqrtf((float)ksub) * 2.0f * sigr * sigr; // C
  }
}

// ---------------------------------------------------------------- WNNM: one wave per 16 x ksub patch
__global__ void __launch_bounds__(256) wnnm_kernel(const float* __restrict__ Eproj,
                                                   float* __restrict__ Eflat,
                                                   const float* __restrict__ scal,
                                                   int ksub) {
  __shared__ float Xs[8][256], Gs[8][256], Vs[8][256], Ms[8][256], Rs[8][16];
  int wave = threadIdx.x >> 5, lane = threadIdx.x & 31;
  int p = blockIdx.x * 8 + wave;           // 0..16383
  int b = p >> 12, pi = p & 4095;
  int hp = pi >> 6, wp = pi & 63;
  float* X = Xs[wave]; float* G = Gs[wave]; float* V = Vs[wave];
  float* M = Ms[wave]; float* rat = Rs[wave];
  const float* Ep = Eproj + (long)b * PADK * HWD;
  const float* mt = scal + 4 + b * PADK;
  float C = scal[69];
  if (lane < 16) { // X: rows=16 pixels, cols=bands (zero-padded)
    int off = (hp * 4 + (lane >> 2)) * WW + (wp * 4 + (lane & 3));
    for (int j = 0; j < 16; ++j)
      X[lane * 16 + j] = (j < ksub) ? (Ep[(long)j * HWD + off] - mt[j]) : 0.0f;
  }
  __syncthreads();
  for (int e = lane; e < 256; e += 32) { // G = X^T X ; V = I
    int i2 = e >> 4, j2 = e & 15;
    float s = 0.0f;
    for (int r = 0; r < 16; ++r) s += X[r * 16 + i2] * X[r * 16 + j2];
    G[e] = s;
    V[e] = (i2 == j2) ? 1.0f : 0.0f;
  }
  __syncthreads();
  if (lane == 0) { // serial cyclic Jacobi on ksub x ksub
    for (int sw = 0; sw < 10; ++sw)
      for (int pp = 0; pp < ksub - 1; ++pp)
        for (int qq = pp + 1; qq < ksub; ++qq) {
          float apq = G[pp * 16 + qq];
          if (fabsf(apq) < 1e-12f) continue;
          float app = G[pp * 16 + pp], aqq = G[qq * 16 + qq];
          float th = (aqq - app) / (2.0f * apq);
          float t = ((th >= 0.0f) ? 1.0f : -1.0f) / (fabsf(th) + sqrtf(th * th + 1.0f));
          float c = 1.0f / sqrtf(t * t + 1.0f), s = t * c;
          for (int r = 0; r < ksub; ++r) {
            float grp = G[r * 16 + pp], grq = G[r * 16 + qq];
            G[r * 16 + pp] = c * grp - s * grq;
            G[r * 16 + qq] = s * grp + c * grq;
          }
          for (int j = 0; j < ksub; ++j) {
            float gpj = G[pp * 16 + j], gqj = G[qq * 16 + j];
            G[pp * 16 + j] = c * gpj - s * gqj;
            G[qq * 16 + j] = s * gpj + c * gqj;
          }
          for (int r = 0; r < ksub; ++r) {
            float vrp = V[r * 16 + pp], vrq = V[r * 16 + qq];
            V[r * 16 + pp] = c * vrp - s * vrq;
            V[r * 16 + qq] = s * vrp + c * vrq;
          }
        }
    for (int j = 0; j < 16; ++j) { // WNNM shrinkage on S = sqrt(eigval)
      float lv = (j < ksub) ? fmaxf(G[j * 16 + j], 0.0f) : 0.0f;
      float S = sqrtf(lv);
      float t2 = (S - 1e-6f) * (S - 1e-6f) - 4.0f * (C - 1e-6f * S);
      float Sn = (t2 > 0.0f) ? fmaxf(S - 1e-6f + sqrtf(fmaxf(t2, 0.0f)), 0.0f) * 0.5f
                             : 0.0f;
      rat[j] = (S > 1e-20f) ? (Sn / S) : 0.0f;
    }
  }
  __syncthreads();
  for (int e = lane; e < 256; e += 32) { // M = V diag(rat) V^T
    int i2 = e >> 4, j2 = e & 15;
    float s = 0.0f;
    for (int t = 0; t < ksub; ++t) s += V[i2 * 16 + t] * rat[t] * V[j2 * 16 + t];
    M[e] = s;
  }
  __syncthreads();
  if (lane < 16) { // Xnew = X*M + mean -> Eflat (image layout)
    int off = (hp * 4 + (lane >> 2)) * WW + (wp * 4 + (lane & 3));
    for (int j = 0; j < 16; ++j) {
      float s = 0.0f;
      for (int t = 0; t < ksub; ++t) s += X[lane * 16 + t] * M[t * 16 + j];
      Eflat[(long)b * PADK * HWD + (long)j * HWD + off] =
          (j < ksub) ? (s + mt[j]) : 0.0f;
    }
  }
}

// ---------------------------------------------------------------- Eimg1 = E @ Eflat, fused blend
__global__ void recon_kernel(const float* __restrict__ E,
                             const float* __restrict__ Eflat,
                             const float* __restrict__ Nimg,
                             const float* __restrict__ inW,
                             const float* __restrict__ scal,
                             float* __restrict__ out, int last) {
  int lane = threadIdx.x & 31;
  int gw = blockIdx.x * (blockDim.x >> 5) + (threadIdx.x >> 5); // 0..32767
  int b = blockIdx.y;
  int tm = gw >> 12, tn = gw & 4095;
  v16bf a = frag_mk(E + (long)b * BAND * 32, 32, tm * 16, 0, lane); // K=0..31, padded
  v16bf bb = frag_kn(Eflat + (long)b * PADK * HWD, HWD, 0, tn * 16, lane, PADK, 1.0f);
  v8f c = wmma_bf16(a, bb, zero8());
  float rb = scal[b];
  int n = tn * 16 + (lane & 15), mh = (lane >> 4) << 3;
#pragma unroll
  for (int r = 0; r < 8; ++r) {
    int m = tm * 16 + r + mh;
    long idx = ((long)b * BAND + m) * HWD + n;
    float e1 = c[r];
    if (last) {
      out[idx] = e1 * (1.0f / 255.0f);
    } else {
      float w = inW[idx];
      float lam = rb / (w + rb);
      out[idx] = (1.0f - lam) * (Nimg[idx] * 255.0f) + lam * e1;
    }
  }
}

// ---------------------------------------------------------------- host
extern "C" void kernel_launch(void* const* d_in, const int* in_sizes, int n_in,
                              void* d_out, int out_size, void* d_ws, size_t ws_size,
                              hipStream_t stream) {
  const float* Nimg = (const float*)d_in[0];
  const float* inW  = (const float*)d_in[1];
  const float* nSig = (const float*)d_in[2];
  float* Eimg = (float*)d_out;                     // Eimg lives in d_out (same size)

  float* ws    = (float*)d_ws;
  float* gram  = ws;                               // 4*128*128
  float* Vg    = gram + NB * BAND * BAND;          // 4*128*128
  float* E     = Vg + NB * BAND * BAND;            // 4*128*32
  float* Eproj = E + NB * BAND * 32;               // 4*16*65536
  float* Nproj = Eproj + NB * PADK * HWD;          // 4*16*65536
  float* Eflat = Nproj + NB * PADK * HWD;          // 4*16*65536
  float* scal  = Eflat + NB * PADK * HWD;          // 128: [0..3]=rho, [4..67]=mean_tE, [68]=sq, [69]=C

  zero_kernel<<<64, 256, 0, stream>>>(scal, 128);
  rho_reduce_kernel<<<dim3(64, NB), 256, 0, stream>>>(inW, scal);
  init_eimg_kernel<<<2048, 256, 0, stream>>>(Nimg, Eimg);
  rho_final_kernel<<<1, 32, 0, stream>>>(scal);

  for (int i = 0; i < 5; ++i) {
    int ksub = 6 + 2 * i;
    zero_kernel<<<256, 256, 0, stream>>>(gram, NB * BAND * BAND);
    gram_kernel<<<dim3(512, NB), 256, 0, stream>>>(Eimg, gram);
    jacobi_kernel<<<NB, 128, 0, stream>>>(gram, Vg, E, ksub);
    proj_kernel<<<dim3(512, NB), 256, 0, stream>>>(E, Eimg, Nimg, Eproj, Nproj);
    zero_kernel<<<1, 256, 0, stream>>>(scal + 4, 66);
    stats_kernel<<<NB * PADK, 256, 0, stream>>>(Eproj, Nproj, scal);
    finalize_kernel<<<1, 32, 0, stream>>>(scal, nSig, ksub, i);
    wnnm_kernel<<<2048, 256, 0, stream>>>(Eproj, Eflat, scal, ksub);
    recon_kernel<<<dim3(4096, NB), 256, 0, stream>>>(E, Eflat, Nimg, inW, scal,
                                                     Eimg, (i == 4) ? 1 : 0);
  }
}
